// BertAttention_12781822673413
// MI455X (gfx1250) — matrix operand order repaired and checked by reading, not compiled, not run on hardware
//
#include <hip/hip_runtime.h>
#include <hip/hip_bf16.h>
#include <stdint.h>

// CDNA5 WMMA types
typedef __attribute__((ext_vector_type(16))) _Float16 v16h;
typedef __attribute__((ext_vector_type(8)))  float    v8f;

#define WMMA_F16(a, b, c) \
  __builtin_amdgcn_wmma_f32_16x16x32_f16(false, (a), false, (b), (short)0, (c), false, false)

// Problem constants
#define BB     4
#define SS     2048
#define DMM    768
#define HH     12
#define HDD    64
#define NSYN_  4
#define SCALE_ 0.125f   // 1/sqrt(64)

#define NT_TILES 48          // DMM/16
#define KSTEPS   24          // DMM/32
#define WPACK    ((size_t)NT_TILES * KSTEPS * 32 * 16)  // 589824 halves per W

union U16 { v16h h; uint32_t u[8]; };

// ---------------------------------------------------------------------------
// Prep 1: convert x (f32 [B*S, DM]) -> f16, straight layout
// ---------------------------------------------------------------------------
__global__ __launch_bounds__(256) void cvt_x_kernel(
    const float* __restrict__ x, _Float16* __restrict__ xh, int n)
{
  for (int i = blockIdx.x * 256 + threadIdx.x; i < n; i += gridDim.x * 256)
    xh[i] = (_Float16)x[i];
}

// ---------------------------------------------------------------------------
// Prep 2: pack the four 768x768 f32 weights into f16 B-fragment order:
//   packed[((nt*KSTEPS + ks)*32 + lane)*16 + h] =
//       W[(nt*16 + (lane&15))*768 + ks*32 + ((lane&16)?16:0) + h]
// so each wave's B fragment is one contiguous, fully coalesced 1KB line.
// ---------------------------------------------------------------------------
__global__ __launch_bounds__(256) void pack_w_kernel(
    const float* __restrict__ W0, const float* __restrict__ W1,
    const float* __restrict__ W2, const float* __restrict__ W3,
    _Float16* __restrict__ wp)
{
  const int w = blockIdx.y;
  const float* W = (w == 0) ? W0 : (w == 1) ? W1 : (w == 2) ? W2 : W3;
  _Float16* out = wp + (size_t)w * WPACK;
  for (size_t p = blockIdx.x * 256 + threadIdx.x; p < WPACK; p += (size_t)gridDim.x * 256) {
    const int h    = (int)(p & 15);
    const int lane = (int)((p >> 4) & 31);
    const int rem  = (int)(p >> 9);
    const int ks   = rem % KSTEPS;
    const int nt   = rem / KSTEPS;
    const int n    = lane & 15;
    const int kbB  = (lane & 16) ? 16 : 0;
    out[p] = (_Float16)W[(size_t)(nt * 16 + n) * DMM + ks * 32 + kbB + h];
  }
}

// ---------------------------------------------------------------------------
// Kernel 1: fused QKV projection (+ bias, + addi K/V injection, q pre-scaled)
//   out = x @ W^T + b ; stored f16 in [B,H,S,HD] layout
// grid: (512 row-tiles, 3 matrices), block: 128 (4 waves)
// ---------------------------------------------------------------------------
__global__ __launch_bounds__(128) void qkv_kernel(
    const _Float16* __restrict__ xh, const _Float16* __restrict__ wpack,
    const float* __restrict__ b0, const float* __restrict__ b1, const float* __restrict__ b2,
    const float* __restrict__ addiK, const float* __restrict__ addiV,
    _Float16* __restrict__ qo, _Float16* __restrict__ ko, _Float16* __restrict__ vo)
{
  __shared__ __align__(16) _Float16 xsh[16 * DMM];   // 24 KB

  const int mt  = blockIdx.x;
  const int mat = blockIdx.y;          // 0=q 1=k 2=v
  const int t   = threadIdx.x;
  const int rowBase = mt * 16;

  // stage x tile (already f16): straight b128 copies
  {
    const uint4* src = (const uint4*)(xh + (size_t)rowBase * DMM);
    uint4* dst = (uint4*)xsh;
    for (int idx = t; idx < (16 * DMM) / 8; idx += 128) dst[idx] = src[idx];
  }
  __syncthreads();

  const _Float16* Wp  = wpack + (size_t)mat * WPACK;
  const float* bias   = (mat == 0) ? b0 : (mat == 1) ? b1 : b2;
  _Float16*    outp   = (mat == 0) ? qo : (mat == 1) ? ko : vo;

  const int wave = t >> 5, lane = t & 31;
  const int mA   = lane & 15;
  const int kbA  = (lane & 16) ? 16 : 0;
  const int nB   = lane & 15;
  const int rbase = (lane & 16) ? 8 : 0;

  for (int nt = wave; nt < NT_TILES; nt += 4) {
    v8f acc = {};
    const int o = nt * 16 + nB;
    const _Float16* wtile = Wp + ((size_t)nt * KSTEPS * 32 + lane) * 16;
    for (int ks = 0; ks < KSTEPS; ++ks) {
      const int kOff = ks * 32;
      U16 A, Bf;
      const uint32_t* p0 = (const uint32_t*)&xsh[mA * DMM + kOff + kbA];
      const uint32_t* p1 = (const uint32_t*)&xsh[mA * DMM + kOff + kbA + 8];
#pragma unroll
      for (int i = 0; i < 4; ++i) { A.u[i] = p0[i]; A.u[4 + i] = p1[i]; }
      const uint32_t* wq = (const uint32_t*)(wtile + (size_t)ks * 32 * 16);
#pragma unroll
      for (int i = 0; i < 8; ++i) Bf.u[i] = wq[i];
      acc = WMMA_F16(A.h, Bf.h, acc);
    }
    const float bval = bias[o];
    const int hh = o >> 6, d = o & 63;
#pragma unroll
    for (int r = 0; r < 8; ++r) {
      const int grow = rowBase + rbase + r;
      const int bidx = grow >> 11, s = grow & (SS - 1);
      float val = acc[r] + bval;
      if (mat == 0) {
        val *= SCALE_;                               // fold softmax scale into q
      } else if (hh < NSYN_) {
        const size_t ai = ((size_t)(bidx * NSYN_ + hh) * SS + s) * HDD + d;
        val += (mat == 1) ? addiK[ai] : addiV[ai];
      }
      outp[((size_t)(bidx * HH + hh) * SS + s) * HDD + d] = (_Float16)val;
    }
  }
}

// ---------------------------------------------------------------------------
// Kernel 2: flash attention. grid: (S/64, B*H), block 128 (4 waves, 16 q each)
// ---------------------------------------------------------------------------
__global__ __launch_bounds__(128) void attn_kernel(
    const _Float16* __restrict__ Q, const _Float16* __restrict__ K,
    const _Float16* __restrict__ V, const float* __restrict__ mask,
    _Float16* __restrict__ ctx)
{
  __shared__ __align__(16) _Float16 Ksh[32 * 64];        // [key][d]
  __shared__ __align__(16) _Float16 Vsh[64 * 32];        // transposed [d][key]
  __shared__ __align__(16) _Float16 Psh[4][16 * 32];     // per-wave P transpose scratch

  const int bh = blockIdx.y;
  const int b = bh / HH, h = bh % HH;
  const _Float16* Qg = Q + (size_t)bh * SS * HDD;
  const _Float16* Kg = K + (size_t)bh * SS * HDD;
  const _Float16* Vg = V + (size_t)bh * SS * HDD;

  const int t = threadIdx.x, wave = t >> 5, lane = t & 31;
  const int qbase = blockIdx.x * 64 + wave * 16;
  const int mA = lane & 15, kbA = (lane & 16) ? 16 : 0;
  const int nB = lane & 15, kbB = (lane & 16) ? 16 : 0;
  const int rbase = (lane & 16) ? 8 : 0;

  // Q fragments (A layout), loaded once; j=0 -> d 0..31, j=1 -> d 32..63
  U16 qa[2];
#pragma unroll
  for (int j = 0; j < 2; ++j) {
    const uint32_t* p0 = (const uint32_t*)&Qg[(size_t)(qbase + mA) * HDD + j * 32 + kbA];
    const uint32_t* p1 = (const uint32_t*)&Qg[(size_t)(qbase + mA) * HDD + j * 32 + kbA + 8];
#pragma unroll
    for (int i = 0; i < 4; ++i) { qa[j].u[i] = p0[i]; qa[j].u[4 + i] = p1[i]; }
  }

  float mrow[8], lrow[8];
  v8f accf[4] = {};
#pragma unroll
  for (int r = 0; r < 8; ++r) { mrow[r] = -1e30f; lrow[r] = 0.f; }

  for (int kblk = 0; kblk < SS; kblk += 32) {
    __syncthreads();
    // stage K block (straight copy, 4 KB)
    for (int idx = t; idx < (32 * 64) / 8; idx += 128)
      ((uint4*)Ksh)[idx] = ((const uint4*)(Kg + (size_t)kblk * HDD))[idx];
    // stage V block transposed
    for (int idx = t; idx < 32 * 64; idx += 128) {
      const int kp = idx >> 6, d = idx & 63;
      Vsh[d * 32 + kp] = Vg[(size_t)(kblk + kp) * HDD + d];
    }
    // prefetch next K/V block while we compute on this one
    if (kblk + 32 < SS && lane == 0) {
      __builtin_prefetch(Kg + (size_t)(kblk + 32) * HDD, 0, 1);
      __builtin_prefetch(Vg + (size_t)(kblk + 32) * HDD, 0, 1);
    }
    __syncthreads();

    // scores: S = q * K^T   (two 16-key column tiles)
    v8f sc[2];
#pragma unroll
    for (int nt2 = 0; nt2 < 2; ++nt2) {
      v8f s = {};
#pragma unroll
      for (int j = 0; j < 2; ++j) {
        U16 Bf;
        const uint32_t* p = (const uint32_t*)&Ksh[(nt2 * 16 + nB) * 64 + j * 32 + kbB];
#pragma unroll
        for (int i = 0; i < 8; ++i) Bf.u[i] = p[i];
        s = WMMA_F16(qa[j].h, Bf.h, s);
      }
      const float mval = mask[(size_t)b * SS + kblk + nt2 * 16 + nB];
#pragma unroll
      for (int r = 0; r < 8; ++r) s[r] += mval;
      sc[nt2] = s;
    }

    // online softmax (row stats reduced across the 16-lane half owning each row)
    float corr[8];
#pragma unroll
    for (int r = 0; r < 8; ++r) {
      float mx = fmaxf(sc[0][r], sc[1][r]);
      mx = fmaxf(mx, __shfl_xor(mx, 1));
      mx = fmaxf(mx, __shfl_xor(mx, 2));
      mx = fmaxf(mx, __shfl_xor(mx, 4));
      mx = fmaxf(mx, __shfl_xor(mx, 8));
      const float mnew = fmaxf(mrow[r], mx);
      const float c  = __expf(mrow[r] - mnew);
      const float p0 = __expf(sc[0][r] - mnew);
      const float p1 = __expf(sc[1][r] - mnew);
      sc[0][r] = p0; sc[1][r] = p1;
      float rs = p0 + p1;
      rs += __shfl_xor(rs, 1);
      rs += __shfl_xor(rs, 2);
      rs += __shfl_xor(rs, 4);
      rs += __shfl_xor(rs, 8);
      lrow[r] = lrow[r] * c + rs;
      mrow[r] = mnew;
      corr[r] = c;
    }
#pragma unroll
    for (int f = 0; f < 4; ++f)
#pragma unroll
      for (int r = 0; r < 8; ++r) accf[f][r] *= corr[r];

    // transpose P (C layout -> A layout) through per-wave LDS scratch
    _Float16* P = Psh[wave];
#pragma unroll
    for (int r = 0; r < 8; ++r) {
      P[(rbase + r) * 32 + nB]      = (_Float16)sc[0][r];
      P[(rbase + r) * 32 + 16 + nB] = (_Float16)sc[1][r];
    }
    __syncthreads();
    U16 pa;
    {
      const uint32_t* p0 = (const uint32_t*)&P[mA * 32 + kbA];
      const uint32_t* p1 = (const uint32_t*)&P[mA * 32 + kbA + 8];
#pragma unroll
      for (int i = 0; i < 4; ++i) { pa.u[i] = p0[i]; pa.u[4 + i] = p1[i]; }
    }

    // ctx += P * V   (4 d-tiles of 16)
#pragma unroll
    for (int f = 0; f < 4; ++f) {
      U16 Bv;
      const uint32_t* p = (const uint32_t*)&Vsh[(f * 16 + nB) * 32 + kbB];
#pragma unroll
      for (int i = 0; i < 8; ++i) Bv.u[i] = p[i];
      accf[f] = WMMA_F16(pa.h, Bv.h, accf[f]);
    }
  }

  // finalize: ctx /= l ; store f16 in [B,S,DM] layout for the O projection
#pragma unroll
  for (int r = 0; r < 8; ++r) lrow[r] = 1.f / lrow[r];
#pragma unroll
  for (int f = 0; f < 4; ++f)
#pragma unroll
    for (int r = 0; r < 8; ++r) {
      const int s = qbase + rbase + r;
      const int d = f * 16 + nB;
      ctx[((size_t)b * SS + s) * DMM + h * HDD + d] = (_Float16)(accf[f][r] * lrow[r]);
    }
}

// ---------------------------------------------------------------------------
// Kernel 3: O projection + bias + residual + LayerNorm. grid 512, block 128.
// ---------------------------------------------------------------------------
__global__ __launch_bounds__(128) void out_ln_kernel(
    const _Float16* __restrict__ ctx, const _Float16* __restrict__ wpackO,
    const float* __restrict__ bo,  const float* __restrict__ x,
    const float* __restrict__ lng, const float* __restrict__ lnb,
    float* __restrict__ out)
{
  __shared__ __align__(16) _Float16 csh[16 * DMM];   // 24 KB
  __shared__ float rowSum[16], rowSq[16];

  const int mt = blockIdx.x, t = threadIdx.x;
  const int rowBase = mt * 16;
  if (t < 16) { rowSum[t] = 0.f; rowSq[t] = 0.f; }
  {
    const uint4* src = (const uint4*)(ctx + (size_t)rowBase * DMM);
    uint4* dst = (uint4*)csh;
    for (int idx = t; idx < (16 * DMM) / 8; idx += 128) dst[idx] = src[idx];
  }
  __syncthreads();

  const int wave = t >> 5, lane = t & 31;
  const int mA = lane & 15, kbA = (lane & 16) ? 16 : 0;
  const int nB = lane & 15;
  const int rbase = (lane & 16) ? 8 : 0;

  float hreg[12][8];
  float psum[8] = {0.f, 0.f, 0.f, 0.f, 0.f, 0.f, 0.f, 0.f};
  float psq[8]  = {0.f, 0.f, 0.f, 0.f, 0.f, 0.f, 0.f, 0.f};

  int ntIdx = 0;
  for (int nt = wave; nt < NT_TILES; nt += 4, ++ntIdx) {
    v8f acc = {};
    const int o = nt * 16 + nB;
    const _Float16* wtile = wpackO + ((size_t)nt * KSTEPS * 32 + lane) * 16;
    for (int ks = 0; ks < KSTEPS; ++ks) {
      const int kOff = ks * 32;
      U16 A, Bf;
      const uint32_t* p0 = (const uint32_t*)&csh[mA * DMM + kOff + kbA];
      const uint32_t* p1 = (const uint32_t*)&csh[mA * DMM + kOff + kbA + 8];
#pragma unroll
      for (int i = 0; i < 4; ++i) { A.u[i] = p0[i]; A.u[4 + i] = p1[i]; }
      const uint32_t* wq = (const uint32_t*)(wtile + (size_t)ks * 32 * 16);
#pragma unroll
      for (int i = 0; i < 8; ++i) Bf.u[i] = wq[i];
      acc = WMMA_F16(A.h, Bf.h, acc);
    }
    const float bval = bo[o];
#pragma unroll
    for (int r = 0; r < 8; ++r) {
      const float hv = acc[r] + bval + x[(size_t)(rowBase + rbase + r) * DMM + o]; // residual
      hreg[ntIdx][r] = hv;
      psum[r] += hv; psq[r] += hv * hv;
    }
  }

  // per-row reduction: shuffle across 16-lane half, then LDS atomics across waves
#pragma unroll
  for (int r = 0; r < 8; ++r) {
    float s = psum[r], q = psq[r];
    s += __shfl_xor(s, 1); q += __shfl_xor(q, 1);
    s += __shfl_xor(s, 2); q += __shfl_xor(q, 2);
    s += __shfl_xor(s, 4); q += __shfl_xor(q, 4);
    s += __shfl_xor(s, 8); q += __shfl_xor(q, 8);
    psum[r] = s; psq[r] = q;
  }
  if ((lane & 15) == 0) {
#pragma unroll
    for (int r = 0; r < 8; ++r) {
      atomicAdd(&rowSum[rbase + r], psum[r]);
      atomicAdd(&rowSq[rbase + r],  psq[r]);
    }
  }
  __syncthreads();

  float mu[8], rs[8];
#pragma unroll
  for (int r = 0; r < 8; ++r) {
    const float m = rowSum[rbase + r] * (1.f / 768.f);
    const float v = rowSq[rbase + r] * (1.f / 768.f) - m * m;
    mu[r] = m; rs[r] = rsqrtf(v + 1e-12f);
  }

  ntIdx = 0;
  for (int nt = wave; nt < NT_TILES; nt += 4, ++ntIdx) {
    const int o = nt * 16 + nB;
    const float g = lng[o], bb = lnb[o];
#pragma unroll
    for (int r = 0; r < 8; ++r)
      out[(size_t)(rowBase + rbase + r) * DMM + o] = (hreg[ntIdx][r] - mu[r]) * rs[r] * g + bb;
  }
}

// ---------------------------------------------------------------------------
extern "C" void kernel_launch(void* const* d_in, const int* in_sizes, int n_in,
                              void* d_out, int out_size, void* d_ws, size_t ws_size,
                              hipStream_t stream) {
  const float* x     = (const float*)d_in[0];
  const float* mask  = (const float*)d_in[1];
  const float* addiK = (const float*)d_in[2];
  const float* addiV = (const float*)d_in[3];
  const float* Wq    = (const float*)d_in[4];
  const float* bq    = (const float*)d_in[5];
  const float* Wk    = (const float*)d_in[6];
  const float* bk    = (const float*)d_in[7];
  const float* Wv    = (const float*)d_in[8];
  const float* bv    = (const float*)d_in[9];
  const float* Wo    = (const float*)d_in[10];
  const float* bo    = (const float*)d_in[11];
  const float* lng   = (const float*)d_in[12];
  const float* lnb   = (const float*)d_in[13];
  float* out = (float*)d_out;

  // workspace: q,k,v [B,H,S,HD] f16 + ctx [B,S,DM] f16 + xh f16 + 4 packed W f16
  const size_t nQKV = (size_t)BB * HH * SS * HDD;   // == B*S*DM = 6291456
  _Float16* qbuf = (_Float16*)d_ws;
  _Float16* kbuf = qbuf + nQKV;
  _Float16* vbuf = kbuf + nQKV;
  _Float16* cbuf = vbuf + nQKV;
  _Float16* xbuf = cbuf + nQKV;
  _Float16* wbuf = xbuf + nQKV;                     // 4 * WPACK halves (q,k,v,o)

  cvt_x_kernel<<<dim3(1024), 256, 0, stream>>>(x, xbuf, (int)nQKV);
  pack_w_kernel<<<dim3(576, 4), 256, 0, stream>>>(Wq, Wk, Wv, Wo, wbuf);

  qkv_kernel<<<dim3((BB * SS) / 16, 3), 128, 0, stream>>>(
      xbuf, wbuf, bq, bk, bv, addiK, addiV, qbuf, kbuf, vbuf);

  attn_kernel<<<dim3(SS / 64, BB * HH), 128, 0, stream>>>(
      qbuf, kbuf, vbuf, mask, cbuf);

  out_ln_kernel<<<dim3((BB * SS) / 16), 128, 0, stream>>>(
      cbuf, wbuf + (size_t)3 * WPACK, bo, x, lng, lnb, out);
}
